// MPNEncoder_78967268704276
// MI455X (gfx1250) — matrix operand order repaired and checked by reading, not compiled, not run on hardware
//
#include <hip/hip_runtime.h>
#include <hip/hip_bf16.h>
#include <stdint.h>

// ---------------------------------------------------------------------------
// MPN encoder (chemprop) for MI455X / gfx1250.
// GEMMs: v_wmma_f32_16x16x32_bf16, fp32 accumulate, bf16 activations.
// Weights staged whole into LDS (<= 306KB of the WGP's 320KB) via
// global_load_async_to_lds_b128 (ASYNCcnt) -> barrier-free K loop.
// ---------------------------------------------------------------------------

#define N_ATOMS   100000
#define N_BONDS   200000
#define MAX_NB    6
#define ATOM_FDIM 133
#define BOND_FDIM 147
#define HIDDEN    300
#define DEPTH     3
#define N_MOLS    4000

#define KP_I 160   // 147 padded to mult of 32
#define KP_H 320   // 300 padded
#define KP_O 448   // 433 padded
#define NPAD 304   // 300 padded to 19*16
#define NT   19    // n-tiles of 16
#define MSG_STRIDE  320  // bond activation row stride (bf16)
#define AMSG_STRIDE 304
#define AIN_STRIDE  448  // concat(f_atoms, a_message), K-ready for W_o
#define HID_STRIDE  304
#define ROWB 36          // LDS row pitch in ushorts (32 data + 4 pad = 72B)

typedef __bf16 v16bf __attribute__((ext_vector_type(16)));
typedef float  v8f   __attribute__((ext_vector_type(8)));
typedef unsigned int v4u __attribute__((ext_vector_type(4)));

union FragAB { v16bf bf; v4u u[2]; };

__device__ __forceinline__ unsigned short f2bf(float f) {
    unsigned int u = __builtin_bit_cast(unsigned int, f);
    u += 0x7FFFu + ((u >> 16) & 1u);          // round-to-nearest-even
    return (unsigned short)(u >> 16);
}
__device__ __forceinline__ float bf2f(unsigned short h) {
    unsigned int u = ((unsigned int)h) << 16;
    return __builtin_bit_cast(float, u);
}

// Async memory->LDS copy, 16B per lane. Tracked by ASYNCcnt.
__device__ __forceinline__ void async_copy_b128(unsigned lds_addr, const void* gsrc) {
    unsigned long long ga = (unsigned long long)(size_t)gsrc;
    asm volatile("global_load_async_to_lds_b128 %0, %1, off"
                 :: "v"(lds_addr), "v"(ga) : "memory");
}
__device__ __forceinline__ void wait_asynccnt0() {
    asm volatile("s_wait_asynccnt 0x0" ::: "memory");
}

// ---------------------------------------------------------------------------
// WMMA GEMM: out[M, 300] = epilogue(A[M, KPAD] @ Wt^T), Wt stored [NPAD][KPAD].
// Block = 256 threads = 8 waves; wave w owns rows [m0, m0+16); each wave keeps
// all 19 n-tile accumulators in VGPRs. Entire Wt staged once into LDS
// (strip-major: KPAD/32 strips of [NPAD rows x 72B]); K loop is barrier-free.
// MODE 0: inp = acc (bf16), msg = relu(acc)
// MODE 1: msg = relu(acc + inp)
// MODE 2: hid = relu(acc + bias[col])  (f32 out)
// ---------------------------------------------------------------------------
template<int KPAD, int MODE>
__global__ __launch_bounds__(256) void k_wmma_gemm(
    const unsigned short* __restrict__ A, int strideA, int M,
    const unsigned short* __restrict__ Wt,
    unsigned short* __restrict__ out_bf,
    unsigned short* __restrict__ inp_bf,
    float* __restrict__ out_f32,
    const float* __restrict__ bias)
{
    constexpr int NSTRIP = KPAD / 32;
    constexpr int STRIPU = NPAD * ROWB;              // ushorts per strip
    __shared__ unsigned short Bsh[NSTRIP * STRIPU];  // up to 306,432 B (KPAD=448)

    const int tid  = threadIdx.x;
    const int lane = tid & 31;
    const int wave = tid >> 5;
    const int lm   = lane & 15;
    const int lh   = lane >> 4;
    const int m0   = blockIdx.x * 128 + wave * 16;

    // ---- stage the whole Wt into LDS with async LDS loads ----
    {
        constexpr int CHUNKS = NSTRIP * NPAD * 4;    // 16B chunks
        const unsigned base = (unsigned)(size_t)(&Bsh[0]);   // LDS byte address
        for (int i = tid; i < CHUNKS; i += 256) {
            int s = i / (NPAD * 4);
            int rem = i - s * (NPAD * 4);
            int r = rem >> 2, c = rem & 3;
            unsigned dst = base + (unsigned)((s * STRIPU + r * ROWB) * 2 + c * 16);
            const unsigned short* src = Wt + (size_t)r * KPAD + s * 32 + c * 8;
            async_copy_b128(dst, src);
        }
        wait_asynccnt0();
    }
    __syncthreads();

    int rowA = m0 + lm; if (rowA >= M) rowA = M - 1;         // clamp; stores predicated
    const unsigned short* aRow = A + (size_t)rowA * (size_t)strideA;

    v8f acc[NT];
    const v8f vzero = {0.f,0.f,0.f,0.f,0.f,0.f,0.f,0.f};
#pragma unroll
    for (int t = 0; t < NT; ++t) acc[t] = vzero;

    for (int s = 0; s < NSTRIP; ++s) {
        // A fragment: lane = row (m0+lm), K = s*32 + 16*lh + {0..15}
        FragAB a;
        a.u[0] = *(const v4u*)(aRow + s * 32 + lh * 16);
        a.u[1] = *(const v4u*)(aRow + s * 32 + lh * 16 + 8);

        const unsigned short* bbase = &Bsh[s * STRIPU + lm * ROWB + lh * 16];
#pragma unroll
        for (int t = 0; t < NT; ++t) {
            // B fragment: lane = Wt row (t*16+lm), K = s*32 + 16*lh + {0..15}
            FragAB b;
            b.u[0] = *(const v4u*)(bbase + t * (16 * ROWB));
            b.u[1] = *(const v4u*)(bbase + t * (16 * ROWB) + 8);
            acc[t] = __builtin_amdgcn_wmma_f32_16x16x32_bf16(
                false, a.bf, false, b.bf, (short)0, acc[t], false, false);
        }
    }

    // epilogue: C/D layout -> element e of lane l is (M = e + 8*lh, N = lm)
#pragma unroll
    for (int t = 0; t < NT; ++t) {
        const int col = t * 16 + lm;
#pragma unroll
        for (int e = 0; e < 8; ++e) {
            int row = m0 + e + 8 * lh;
            if (row >= M) continue;
            float v = acc[t][e];
            if constexpr (MODE == 0) {
                inp_bf[(size_t)row * MSG_STRIDE + col] = f2bf(v);
                out_bf[(size_t)row * MSG_STRIDE + col] = f2bf(v > 0.f ? v : 0.f);
            } else if constexpr (MODE == 1) {
                v += bf2f(inp_bf[(size_t)row * MSG_STRIDE + col]);
                out_bf[(size_t)row * MSG_STRIDE + col] = f2bf(v > 0.f ? v : 0.f);
            } else {
                int bc = col < (HIDDEN - 1) ? col : (HIDDEN - 1);
                v += bias[bc];
                out_f32[(size_t)row * HID_STRIDE + col] = v > 0.f ? v : 0.f;
            }
        }
    }
}

// --- prep kernels -----------------------------------------------------------
__global__ void k_prep_w(const float* __restrict__ W, unsigned short* __restrict__ Wt,
                         int K, int Kpad) {
    int idx = blockIdx.x * blockDim.x + threadIdx.x;
    if (idx >= K * HIDDEN) return;
    int k = idx / HIDDEN, n = idx % HIDDEN;
    Wt[(size_t)n * (size_t)Kpad + k] = f2bf(W[idx]);   // transpose + cvt; pad stays 0
}

__global__ void k_prep_fbonds(const float* __restrict__ fb, unsigned short* __restrict__ o) {
    int idx = blockIdx.x * blockDim.x + threadIdx.x;
    if (idx >= N_BONDS * BOND_FDIM) return;
    int b = idx / BOND_FDIM, k = idx % BOND_FDIM;
    o[(size_t)b * KP_I + k] = f2bf(fb[idx]);
}

__global__ void k_prep_fatoms(const float* __restrict__ fa, unsigned short* __restrict__ ain) {
    int idx = blockIdx.x * blockDim.x + threadIdx.x;
    if (idx >= N_ATOMS * 148) return;                  // 133 data + 15 zero-pad cols
    int a = idx / 148, j = idx % 148;
    if (j < ATOM_FDIM)
        ain[(size_t)a * AIN_STRIDE + j] = f2bf(fa[(size_t)a * ATOM_FDIM + j]);
    else
        ain[(size_t)a * AIN_STRIDE + 433 + (j - ATOM_FDIM)] = 0;  // cols 433..447
}

// --- gather kernels ---------------------------------------------------------
__global__ __launch_bounds__(320) void k_amsg(
    const unsigned short* __restrict__ msg, const int* __restrict__ a2b,
    unsigned short* __restrict__ dst, int dstStride, int dstOff)
{
    int a = blockIdx.x, c = threadIdx.x;
    if (c >= HIDDEN) return;
    float s = 0.f;
#pragma unroll
    for (int j = 0; j < MAX_NB; ++j) {
        int b = a2b[a * MAX_NB + j];
        s += bf2f(msg[(size_t)b * MSG_STRIDE + c]);
    }
    dst[(size_t)a * (size_t)dstStride + dstOff + c] = f2bf(s);
}

__global__ __launch_bounds__(320) void k_tmp(
    const unsigned short* __restrict__ amsg, const unsigned short* __restrict__ msg,
    const int* __restrict__ b2a, const int* __restrict__ b2revb,
    unsigned short* __restrict__ tmp)
{
    int b = blockIdx.x, c = threadIdx.x;
    if (c >= HIDDEN) return;
    float v = bf2f(amsg[(size_t)b2a[b] * AMSG_STRIDE + c])
            - bf2f(msg[(size_t)b2revb[b] * MSG_STRIDE + c]);
    tmp[(size_t)b * MSG_STRIDE + c] = f2bf(v);
}

// --- per-molecule mean over sorted mol_ids ---------------------------------
__global__ __launch_bounds__(320) void k_segmean(
    const float* __restrict__ hid, const int* __restrict__ mol_ids,
    float* __restrict__ out)
{
    __shared__ int s_lo, s_hi;
    int m = blockIdx.x, c = threadIdx.x;
    if (c == 0) {
        int lo = 0, hi = N_ATOMS;
        while (lo < hi) { int mid = (lo + hi) >> 1; if (mol_ids[mid] <  m) lo = mid + 1; else hi = mid; }
        s_lo = lo;
        hi = N_ATOMS;
        while (lo < hi) { int mid = (lo + hi) >> 1; if (mol_ids[mid] <= m) lo = mid + 1; else hi = mid; }
        s_hi = lo;
    }
    __syncthreads();
    if (c >= HIDDEN) return;
    int lo = s_lo, hi = s_hi;
    float s = 0.f;
    for (int r = lo; r < hi; ++r) s += hid[(size_t)r * HID_STRIDE + c];
    int cnt = hi - lo;
    out[(size_t)m * HIDDEN + c] = cnt > 0 ? (s / (float)cnt) : 0.f;
}

// ---------------------------------------------------------------------------
extern "C" void kernel_launch(void* const* d_in, const int* in_sizes, int n_in,
                              void* d_out, int out_size, void* d_ws, size_t ws_size,
                              hipStream_t stream) {
    (void)in_sizes; (void)n_in; (void)out_size; (void)ws_size;
    const float* f_atoms = (const float*)d_in[0];
    const float* f_bonds = (const float*)d_in[1];
    const int*   a2b     = (const int*)d_in[2];
    const int*   b2a     = (const int*)d_in[3];
    const int*   b2revb  = (const int*)d_in[4];
    const int*   mol_ids = (const int*)d_in[5];
    const float* W_i     = (const float*)d_in[6];
    const float* W_h     = (const float*)d_in[7];
    const float* W_o     = (const float*)d_in[8];
    const float* b_o     = (const float*)d_in[9];
    float* out = (float*)d_out;

    char* ws = (char*)d_ws;
    const size_t OFF_INP  = 0;                                          // bf16 [N_BONDS][320]; later f32 hidden
    const size_t OFF_MSG  = OFF_INP  + (size_t)N_BONDS * MSG_STRIDE * 2;
    const size_t OFF_TMP  = OFF_MSG  + (size_t)N_BONDS * MSG_STRIDE * 2; // later aliased as a_input
    const size_t OFF_AMSG = OFF_TMP  + (size_t)N_BONDS * MSG_STRIDE * 2;
    const size_t OFF_FB   = OFF_AMSG + (size_t)N_ATOMS * AMSG_STRIDE * 2;
    const size_t OFF_WTI  = OFF_FB   + (size_t)N_BONDS * KP_I * 2;
    const size_t OFF_WTH  = OFF_WTI  + (size_t)NPAD * KP_I * 2;
    const size_t OFF_WTO  = OFF_WTH  + (size_t)NPAD * KP_H * 2;
    const size_t TOTAL    = OFF_WTO  + (size_t)NPAD * KP_O * 2;

    unsigned short* inp    = (unsigned short*)(ws + OFF_INP);
    unsigned short* msg    = (unsigned short*)(ws + OFF_MSG);
    unsigned short* tmp    = (unsigned short*)(ws + OFF_TMP);
    unsigned short* ain    = tmp;                       // alias: tmp dead before a_input built
    unsigned short* amsg   = (unsigned short*)(ws + OFF_AMSG);
    unsigned short* fb16   = (unsigned short*)(ws + OFF_FB);
    unsigned short* wti    = (unsigned short*)(ws + OFF_WTI);
    unsigned short* wth    = (unsigned short*)(ws + OFF_WTH);
    unsigned short* wto    = (unsigned short*)(ws + OFF_WTO);
    float*          hidden = (float*)(ws + OFF_INP);    // alias: inp dead after last W_h GEMM

    hipMemsetAsync(d_ws, 0, TOTAL, stream);             // zero-fill (capture-safe)

    k_prep_w<<<(BOND_FDIM * HIDDEN + 255) / 256, 256, 0, stream>>>(W_i, wti, BOND_FDIM, KP_I);
    k_prep_w<<<(HIDDEN * HIDDEN + 255) / 256, 256, 0, stream>>>(W_h, wth, HIDDEN, KP_H);
    k_prep_w<<<((ATOM_FDIM + HIDDEN) * HIDDEN + 255) / 256, 256, 0, stream>>>(W_o, wto, ATOM_FDIM + HIDDEN, KP_O);
    k_prep_fbonds<<<(N_BONDS * BOND_FDIM + 255) / 256, 256, 0, stream>>>(f_bonds, fb16);

    const int gb = (N_BONDS + 127) / 128;
    const int ga = (N_ATOMS + 127) / 128;

    // inp = f_bonds @ W_i ; msg = relu(inp)
    k_wmma_gemm<KP_I, 0><<<gb, 256, 0, stream>>>(fb16, KP_I, N_BONDS, wti,
                                                 msg, inp, nullptr, nullptr);
    for (int d = 0; d < DEPTH - 1; ++d) {
        k_amsg<<<N_ATOMS, 320, 0, stream>>>(msg, a2b, amsg, AMSG_STRIDE, 0);
        k_tmp<<<N_BONDS, 320, 0, stream>>>(amsg, msg, b2a, b2revb, tmp);
        // msg = relu(inp + tmp @ W_h)
        k_wmma_gemm<KP_H, 1><<<gb, 256, 0, stream>>>(tmp, MSG_STRIDE, N_BONDS, wth,
                                                     msg, inp, nullptr, nullptr);
    }
    // a_input = concat(f_atoms, sum_nb msg)  (bf16, K padded to 448 w/ zeros)
    k_prep_fatoms<<<(N_ATOMS * 148 + 255) / 256, 256, 0, stream>>>(f_atoms, ain);
    k_amsg<<<N_ATOMS, 320, 0, stream>>>(msg, a2b, ain, AIN_STRIDE, ATOM_FDIM);
    // hidden = relu(a_input @ W_o + b_o)
    k_wmma_gemm<KP_O, 2><<<ga, 256, 0, stream>>>(ain, AIN_STRIDE, N_ATOMS, wto,
                                                 nullptr, nullptr, hidden, b_o);
    // per-molecule mean
    k_segmean<<<N_MOLS, 320, 0, stream>>>(hidden, mol_ids, out);
}
